// EGNNRegressor_42795054138026
// MI455X (gfx1250) — compile-verified, matrix operand order checked by reading, not run on hardware
//
#include <hip/hip_runtime.h>
#include <hip/hip_bf16.h>
#include <cstdint>
#include <cstddef>

// ---------------------------------------------------------------------------
// EGNN regressor for MI455X (gfx1250). All linear layers run as wave32 WMMA
// f32<-f16xf16 (v_wmma_f32_16x16x32_f16). Edge MLP chain fully fused in LDS,
// with the h[dst]/h[src] gather staged via GLOBAL_LOAD_ASYNC_TO_LDS_B128
// (ASYNCcnt) so the gather bypasses VGPRs entirely.
// ---------------------------------------------------------------------------

typedef __attribute__((ext_vector_type(16))) _Float16 v16h;
typedef __attribute__((ext_vector_type(8)))  _Float16 v8h;
typedef __attribute__((ext_vector_type(8)))  float    v8f;

namespace {
constexpr int NN = 20000;   // nodes
constexpr int EE = 320000;  // edges
constexpr int GG = 64;      // graphs
constexpr int HH = 128;     // hidden
constexpr int NLAYERS = 4;
constexpr int K_EMB  = 96;   // 70 padded to 96  (mult of 32)
constexpr int K_EDGE = 288;  // 258 padded to 288
constexpr int K_CAT  = 256;  // 2*H
}

__device__ __forceinline__ float siluf(float x) { return x / (1.0f + __expf(-x)); }

// Load one 16x32 f16 WMMA fragment (A row-tile or B col-tile from a
// pre-transposed [Nout x Kpad] weight). Per ISA 7.12.2, lane (l%16) holds
// row/col l%16; halves 0..7 hold K = 8*(l/16)+0..7, halves 8..15 hold
// K = 16 + 8*(l/16) + 0..7  -> two contiguous 16B loads per lane.
__device__ __forceinline__ v16h load_frag(const _Float16* base, int ld, int row,
                                          int k0, int half) {
  const _Float16* p = base + (size_t)row * ld + (k0 + half * 8);
  v8h lo = *(const v8h*)(p);
  v8h hi = *(const v8h*)(p + 16);
  return __builtin_shufflevector(lo, hi, 0,1,2,3,4,5,6,7,8,9,10,11,12,13,14,15);
}

__device__ __forceinline__ v8f wmma_f16(v16h a, v16h b, v8f c) {
  return __builtin_amdgcn_wmma_f32_16x16x32_f16(false, a, false, b, (short)0, c,
                                                false, false);
}

// ---------------------------------------------------------------------------
// Generic wave-tiled GEMM: out[M x 128] = act(A[M x K](f16) @ Wt^T + bias)
// Wt is f16 [128 x Kpad] (pre-transposed weights). Each wave produces a
// 16x128 strip: the A fragment is loaded once per K-step and reused across
// all 8 column tiles (8 wmma per A fragment).
// ---------------------------------------------------------------------------
__global__ void gemm16_kernel(const _Float16* __restrict__ A, int lda,
                              const _Float16* __restrict__ Wt, int ldb,
                              const float* __restrict__ bias,
                              float* __restrict__ outF, _Float16* __restrict__ outH,
                              int M, int K, int act) {
  const int lane = threadIdx.x & 31, wave = threadIdx.x >> 5;
  const int m0 = blockIdx.x * 64 + wave * 16;
  if (m0 >= M) return;
  const int half = lane >> 4, r = lane & 15;
  v8f acc[8];
#pragma unroll
  for (int nc = 0; nc < 8; ++nc) {
    const float bv = bias ? bias[nc * 16 + r] : 0.0f;
    acc[nc] = (v8f){bv, bv, bv, bv, bv, bv, bv, bv};
  }
  for (int k0 = 0; k0 < K; k0 += 32) {
    const v16h a = load_frag(A, lda, m0 + r, k0, half);
#pragma unroll
    for (int nc = 0; nc < 8; ++nc) {
      v16h b = load_frag(Wt, ldb, nc * 16 + r, k0, half);
      acc[nc] = wmma_f16(a, b, acc[nc]);
    }
  }
#pragma unroll
  for (int nc = 0; nc < 8; ++nc) {
#pragma unroll
    for (int j = 0; j < 8; ++j) {
      const int row = m0 + j + half * 8;
      const int col = nc * 16 + r;
      float v = acc[nc][j];
      if (act == 1) v = siluf(v);
      else if (act == 2) v = tanhf(v);
      if (outF) outF[(size_t)row * HH + col] = v;
      if (outH) outH[(size_t)row * HH + col] = (_Float16)v;
    }
  }
}

// ---------------------------------------------------------------------------
// Fused per-layer edge kernel. 128 edges per block, 256 threads (8 waves).
//   m_in = [h[dst] | h[src] | r2 | ea | 0pad]  (LDS, f16, 128x288)
//   t1   = silu(m_in @ e1)         t1: 128x128  (LDS)
//   m_ij = silu(t1 @ e2)           + atomic agg[dst] += m_ij
//   t2   = silu(m_ij @ x1)
//   gate = tanh(t2 @ x2);  posacc[dst] += diff * gate / (r2+1)
// Gather of h rows goes memory->LDS via global_load_async_to_lds_b128.
// ---------------------------------------------------------------------------
__global__ void egnn_edge_kernel(const int* __restrict__ ei,           // [2,E]
                                 const float* __restrict__ eattr,      // [E]
                                 const float* __restrict__ posc,       // [N,3]
                                 const _Float16* __restrict__ hf16,    // [N,128]
                                 const _Float16* __restrict__ e1t, const float* __restrict__ e1b,
                                 const _Float16* __restrict__ e2t, const float* __restrict__ e2b,
                                 const _Float16* __restrict__ x1t, const float* __restrict__ x1b,
                                 const float* __restrict__ x2w, const float* __restrict__ x2b,
                                 float* __restrict__ agg,              // [N,128]
                                 float* __restrict__ posacc) {         // [N,3]
  extern __shared__ unsigned char smem_raw[];
  _Float16* sA = (_Float16*)smem_raw;          // 128 x 288
  _Float16* sT = sA + 128 * K_EDGE;            // 128 x 128
  _Float16* sM = sT + 128 * HH;                // 128 x 128
  float* sDiff = (float*)(sM + 128 * HH);      // 128 x 3
  float* sR2   = sDiff + 128 * 3;              // 128
  int*   sDst  = (int*)(sR2 + 128);            // 128

  const int tid = threadIdx.x;
  const int e0  = blockIdx.x * 128;
  {
    const int e = tid >> 1, part = tid & 1;
    const int ge  = e0 + e;
    const int src = ei[ge];
    const int dst = ei[EE + ge];
    if (part == 0) {
      float dx = posc[dst * 3 + 0] - posc[src * 3 + 0];
      float dy = posc[dst * 3 + 1] - posc[src * 3 + 1];
      float dz = posc[dst * 3 + 2] - posc[src * 3 + 2];
      sDiff[e * 3 + 0] = dx; sDiff[e * 3 + 1] = dy; sDiff[e * 3 + 2] = dz;
      sR2[e]  = dx * dx + dy * dy + dz * dz;
      sDst[e] = dst;
    }
    // Async gather: 16B chunks of h row, memory -> LDS (no VGPR round trip).
    const _Float16* hrow = hf16 + (size_t)(part ? src : dst) * HH;
    _Float16* arow = sA + e * K_EDGE + part * HH;
#pragma unroll
    for (int c = 0; c < HH; c += 8) {
      const unsigned lds_off = (unsigned)(uintptr_t)(arow + c);          // LDS byte offset (addr[31:0])
      const unsigned long long ga = (unsigned long long)(uintptr_t)(hrow + c);
      asm volatile("global_load_async_to_lds_b128 %0, %1, off"
                   :: "v"(lds_off), "v"(ga) : "memory");
    }
  }
  asm volatile("s_wait_asynccnt 0x0" ::: "memory");
  __syncthreads();
  if (tid < 128) {
    _Float16* rowp = sA + tid * K_EDGE;
    rowp[256] = (_Float16)sR2[tid];
    rowp[257] = (_Float16)eattr[e0 + tid];
#pragma unroll
    for (int c = 258; c < K_EDGE; ++c) rowp[c] = (_Float16)0.0f;
  }
  __syncthreads();

  const int lane = tid & 31, wave = tid >> 5;
  const int half = lane >> 4, r = lane & 15;
  const int m0 = wave * 16;  // 8 waves x 16 edges = 128

  // GEMM1: t1 = silu(sA @ e1 + e1b), K = 288. A frags register-cached.
  {
    v16h afr[K_EDGE / 32];
#pragma unroll
    for (int kt = 0; kt < K_EDGE / 32; ++kt)
      afr[kt] = load_frag(sA, K_EDGE, m0 + r, kt * 32, half);
    for (int nc = 0; nc < 8; ++nc) {
      const int n0 = nc * 16;
      const float bv = e1b[n0 + r];
      v8f acc = {bv, bv, bv, bv, bv, bv, bv, bv};
#pragma unroll
      for (int kt = 0; kt < K_EDGE / 32; ++kt) {
        v16h b = load_frag(e1t, K_EDGE, n0 + r, kt * 32, half);
        acc = wmma_f16(afr[kt], b, acc);
      }
#pragma unroll
      for (int j = 0; j < 8; ++j)
        sT[(m0 + j + half * 8) * HH + n0 + r] = (_Float16)siluf(acc[j]);
    }
  }
  __syncthreads();

  // GEMM2: m_ij = silu(t1 @ e2 + e2b); scatter agg
  {
    v16h afr[HH / 32];
#pragma unroll
    for (int kt = 0; kt < HH / 32; ++kt)
      afr[kt] = load_frag(sT, HH, m0 + r, kt * 32, half);
    for (int nc = 0; nc < 8; ++nc) {
      const int n0 = nc * 16;
      const float bv = e2b[n0 + r];
      v8f acc = {bv, bv, bv, bv, bv, bv, bv, bv};
#pragma unroll
      for (int kt = 0; kt < HH / 32; ++kt) {
        v16h b = load_frag(e2t, HH, n0 + r, kt * 32, half);
        acc = wmma_f16(afr[kt], b, acc);
      }
#pragma unroll
      for (int j = 0; j < 8; ++j) {
        const int rowe = m0 + j + half * 8;
        const float v = siluf(acc[j]);
        sM[rowe * HH + n0 + r] = (_Float16)v;
        atomicAdd(&agg[(size_t)sDst[rowe] * HH + n0 + r], v);
      }
    }
  }
  __syncthreads();

  // GEMM3: t2 = silu(m_ij @ x1 + x1b) -> sT (reuse)
  {
    v16h afr[HH / 32];
#pragma unroll
    for (int kt = 0; kt < HH / 32; ++kt)
      afr[kt] = load_frag(sM, HH, m0 + r, kt * 32, half);
    for (int nc = 0; nc < 8; ++nc) {
      const int n0 = nc * 16;
      const float bv = x1b[n0 + r];
      v8f acc = {bv, bv, bv, bv, bv, bv, bv, bv};
#pragma unroll
      for (int kt = 0; kt < HH / 32; ++kt) {
        v16h b = load_frag(x1t, HH, n0 + r, kt * 32, half);
        acc = wmma_f16(afr[kt], b, acc);
      }
#pragma unroll
      for (int j = 0; j < 8; ++j)
        sT[(m0 + j + half * 8) * HH + n0 + r] = (_Float16)siluf(acc[j]);
    }
  }
  __syncthreads();

  // gate + position accumulation (matvec 128 -> 1)
  if (tid < 128) {
    float dot = x2b[0];
#pragma unroll 8
    for (int k = 0; k < HH; ++k) dot += (float)sT[tid * HH + k] * x2w[k];
    const float gate = tanhf(dot);
    const float coef = gate / (sR2[tid] + 1.0f);
    const int d = sDst[tid];
    atomicAdd(&posacc[d * 3 + 0], sDiff[tid * 3 + 0] * coef);
    atomicAdd(&posacc[d * 3 + 1], sDiff[tid * 3 + 1] * coef);
    atomicAdd(&posacc[d * 3 + 2], sDiff[tid * 3 + 2] * coef);
  }
}

// ------------------------------ small kernels ------------------------------
__global__ void zero_f32_kernel(float* __restrict__ p, int n) {
  int i = blockIdx.x * blockDim.x + threadIdx.x;
  if (i < n) p[i] = 0.0f;
}

__global__ void transpose_w_kernel(const float* __restrict__ W, _Float16* __restrict__ Wt,
                                   int K, int Nout, int Kpad) {
  int idx = blockIdx.x * blockDim.x + threadIdx.x;
  if (idx >= Nout * Kpad) return;
  int n = idx / Kpad, k = idx % Kpad;
  Wt[idx] = (k < K) ? (_Float16)W[(size_t)k * Nout + n] : (_Float16)0.0f;
}

__global__ void scatter_pos_kernel(const float* __restrict__ pos, const int* __restrict__ batch,
                                   float* __restrict__ possum, float* __restrict__ cnt, int n) {
  int i = blockIdx.x * blockDim.x + threadIdx.x;
  if (i >= n) return;
  int b = batch[i];
  atomicAdd(&possum[b * 3 + 0], pos[i * 3 + 0]);
  atomicAdd(&possum[b * 3 + 1], pos[i * 3 + 1]);
  atomicAdd(&possum[b * 3 + 2], pos[i * 3 + 2]);
  atomicAdd(&cnt[b], 1.0f);
}

__global__ void center_pos_kernel(const float* __restrict__ pos, const int* __restrict__ batch,
                                  const float* __restrict__ possum, const float* __restrict__ cnt,
                                  float* __restrict__ posc, int n) {
  int i = blockIdx.x * blockDim.x + threadIdx.x;
  if (i >= n) return;
  int b = batch[i];
  float c = fmaxf(cnt[b], 1.0f);
  posc[i * 3 + 0] = pos[i * 3 + 0] - possum[b * 3 + 0] / c;
  posc[i * 3 + 1] = pos[i * 3 + 1] - possum[b * 3 + 1] / c;
  posc[i * 3 + 2] = pos[i * 3 + 2] - possum[b * 3 + 2] / c;
}

__global__ void build_ain_kernel(const int* __restrict__ z, const float* __restrict__ x,
                                 const float* __restrict__ zemb, _Float16* __restrict__ ain) {
  int i = blockIdx.x, t = threadIdx.x;  // 96 threads per node
  float v = 0.0f;
  if (t < 64) v = zemb[(size_t)z[i] * 64 + t];
  else if (t < 70) v = x[(size_t)i * 6 + (t - 64)];
  ain[(size_t)i * K_EMB + t] = (_Float16)v;
}

__global__ void pack_hcat_kernel(const _Float16* __restrict__ hf16, float* __restrict__ agg,
                                 _Float16* __restrict__ hcat, int n) {
  int idx = blockIdx.x * blockDim.x + threadIdx.x;
  if (idx >= n * 256) return;
  int i = idx >> 8, c = idx & 255;
  if (c < HH) {
    hcat[idx] = hf16[(size_t)i * HH + c];
  } else {
    hcat[idx] = (_Float16)agg[(size_t)i * HH + (c - HH)];
    agg[(size_t)i * HH + (c - HH)] = 0.0f;  // ready for next layer
  }
}

__global__ void layernorm_kernel(float* __restrict__ h, const float* __restrict__ dh,
                                 const float* __restrict__ g, const float* __restrict__ b,
                                 _Float16* __restrict__ hf16) {
  __shared__ float red[HH];
  const int i = blockIdx.x, t = threadIdx.x;
  float v = h[(size_t)i * HH + t] + dh[(size_t)i * HH + t];
  red[t] = v; __syncthreads();
  for (int s = 64; s > 0; s >>= 1) { if (t < s) red[t] += red[t + s]; __syncthreads(); }
  const float mean = red[0] / (float)HH; __syncthreads();
  const float c = v - mean;
  red[t] = c * c; __syncthreads();
  for (int s = 64; s > 0; s >>= 1) { if (t < s) red[t] += red[t + s]; __syncthreads(); }
  const float var = red[0] / (float)HH;
  const float o = c * rsqrtf(var + 1e-5f) * g[t] + b[t];
  h[(size_t)i * HH + t] = o;
  hf16[(size_t)i * HH + t] = (_Float16)o;
}

__global__ void pos_apply_kernel(float* __restrict__ posc, float* __restrict__ posacc, int n3) {
  int i = blockIdx.x * blockDim.x + threadIdx.x;
  if (i >= n3) return;
  posc[i] += posacc[i];
  posacc[i] = 0.0f;
}

__global__ void scatter_h_kernel(const float* __restrict__ h, const int* __restrict__ batch,
                                 float* __restrict__ hgsum) {
  int i = blockIdx.x, t = threadIdx.x;  // 128 threads per node
  atomicAdd(&hgsum[(size_t)batch[i] * HH + t], h[(size_t)i * HH + t]);
}

__global__ void hg_finalize_kernel(const float* __restrict__ hgsum, const float* __restrict__ cnt,
                                   _Float16* __restrict__ hg16) {
  int idx = blockIdx.x * blockDim.x + threadIdx.x;
  if (idx >= GG * HH) return;
  int g = idx >> 7;
  hg16[idx] = (_Float16)(hgsum[idx] / fmaxf(cnt[g], 1.0f));
}

__global__ void readout_final_kernel(const _Float16* __restrict__ tro,
                                     const float* __restrict__ ro2w,
                                     const float* __restrict__ ro2b,
                                     float* __restrict__ out) {
  int g = threadIdx.x;
  if (g >= GG) return;
  float dot = ro2b[0];
#pragma unroll 8
  for (int k = 0; k < HH; ++k) dot += (float)tro[(size_t)g * HH + k] * ro2w[k];
  out[g] = dot;
}

// ---------------------------------------------------------------------------
// Host side
// ---------------------------------------------------------------------------
extern "C" void kernel_launch(void* const* d_in, const int* in_sizes, int n_in,
                              void* d_out, int out_size, void* d_ws, size_t ws_size,
                              hipStream_t stream) {
  (void)in_sizes; (void)n_in; (void)out_size; (void)ws_size;
  // Input order: top-level dict insertion order, params pytree-flattened
  // (sorted keys: in1,in2,layers[*]{e1,e2,h1,h2,ln_b,ln_g,x1,x2},ro1,ro2,z_emb;
  //  each linear as b,w).
  const int*   z      = (const int*)d_in[0];
  const float* x      = (const float*)d_in[1];
  const float* pos    = (const float*)d_in[2];
  const int*   eidx   = (const int*)d_in[3];
  const float* eattr  = (const float*)d_in[4];
  const int*   batch  = (const int*)d_in[5];
  const float* in1b = (const float*)d_in[7],  *in1w = (const float*)d_in[8];
  const float* in2b = (const float*)d_in[9],  *in2w = (const float*)d_in[10];
  auto LP = [&](int l, int off) { return (const float*)d_in[11 + l * 14 + off]; };
  // per-layer offsets: e1b=0,e1w=1,e2b=2,e2w=3,h1b=4,h1w=5,h2b=6,h2w=7,
  //                    lnb=8,lng=9,x1b=10,x1w=11,x2b=12,x2w=13
  const float* ro1b = (const float*)d_in[67], *ro1w = (const float*)d_in[68];
  const float* ro2b = (const float*)d_in[69], *ro2w = (const float*)d_in[70];
  const float* zemb = (const float*)d_in[71];
  float* out = (float*)d_out;

  // ---- workspace carve (total ~57 MB) ----
  size_t cur = 0;
  auto alloc = [&](size_t bytes) -> char* {
    cur = (cur + 255) & ~(size_t)255;
    char* p = (char*)d_ws + cur;
    cur += bytes;
    return p;
  };
  float*    posc   = (float*)alloc((size_t)NN * 3 * 4);
  float*    posacc = (float*)alloc((size_t)NN * 3 * 4);
  float*    possum = (float*)alloc((size_t)GG * 3 * 4);
  float*    cnt    = (float*)alloc((size_t)GG * 4);
  float*    h      = (float*)alloc((size_t)NN * HH * 4);
  _Float16* hf16   = (_Float16*)alloc((size_t)NN * HH * 2);
  float*    agg    = (float*)alloc((size_t)NN * HH * 4);
  _Float16* hcat   = (_Float16*)alloc((size_t)NN * K_CAT * 2);
  _Float16* tnode  = (_Float16*)alloc((size_t)NN * HH * 2);
  float*    dh     = (float*)alloc((size_t)NN * HH * 4);
  _Float16* ain    = (_Float16*)alloc((size_t)NN * K_EMB * 2);
  float*    hgsum  = (float*)alloc((size_t)GG * HH * 4);
  _Float16* hg16   = (_Float16*)alloc((size_t)GG * HH * 2);
  _Float16* tro    = (_Float16*)alloc((size_t)GG * HH * 2);
  _Float16* in1t   = (_Float16*)alloc((size_t)HH * K_EMB * 2);
  _Float16* in2t   = (_Float16*)alloc((size_t)HH * HH * 2);
  _Float16* ro1t   = (_Float16*)alloc((size_t)HH * HH * 2);
  _Float16 *e1t[NLAYERS], *e2t[NLAYERS], *h1t[NLAYERS], *h2t[NLAYERS], *x1t[NLAYERS];
  for (int l = 0; l < NLAYERS; ++l) {
    e1t[l] = (_Float16*)alloc((size_t)HH * K_EDGE * 2);
    e2t[l] = (_Float16*)alloc((size_t)HH * HH * 2);
    h1t[l] = (_Float16*)alloc((size_t)HH * K_CAT * 2);
    h2t[l] = (_Float16*)alloc((size_t)HH * HH * 2);
    x1t[l] = (_Float16*)alloc((size_t)HH * HH * 2);
  }

  auto zero = [&](float* p, int n) {
    zero_f32_kernel<<<(n + 255) / 256, 256, 0, stream>>>(p, n);
  };
  auto transpose = [&](const float* W, _Float16* Wt, int K, int Nout, int Kpad) {
    transpose_w_kernel<<<(Nout * Kpad + 255) / 256, 256, 0, stream>>>(W, Wt, K, Nout, Kpad);
  };
  auto gemm = [&](const _Float16* A, int lda, const _Float16* Wt, int ldb,
                  const float* bias, float* oF, _Float16* oH, int M, int K, int act) {
    gemm16_kernel<<<(M + 63) / 64, 128, 0, stream>>>(A, lda, Wt, ldb, bias, oF, oH, M, K, act);
  };

  // ---- init accumulators ----
  zero(posacc, NN * 3); zero(possum, GG * 3); zero(cnt, GG);
  zero(agg, NN * HH);   zero(hgsum, GG * HH);

  // ---- weight transpose + f16 convert (stay L2-resident thereafter) ----
  transpose(in1w, in1t, 70, HH, K_EMB);
  transpose(in2w, in2t, HH, HH, HH);
  transpose(ro1w, ro1t, HH, HH, HH);
  for (int l = 0; l < NLAYERS; ++l) {
    transpose(LP(l, 1),  e1t[l], 258, HH, K_EDGE);
    transpose(LP(l, 3),  e2t[l], HH,  HH, HH);
    transpose(LP(l, 5),  h1t[l], K_CAT, HH, K_CAT);
    transpose(LP(l, 7),  h2t[l], HH,  HH, HH);
    transpose(LP(l, 11), x1t[l], HH,  HH, HH);
  }

  // ---- center positions per graph ----
  scatter_pos_kernel<<<(NN + 255) / 256, 256, 0, stream>>>(pos, batch, possum, cnt, NN);
  center_pos_kernel<<<(NN + 255) / 256, 256, 0, stream>>>(pos, batch, possum, cnt, posc, NN);

  // ---- node embedding: h = in2(silu(in1([zemb|x]))) ----
  build_ain_kernel<<<NN, K_EMB, 0, stream>>>(z, x, zemb, ain);
  gemm(ain, K_EMB, in1t, K_EMB, in1b, nullptr, tnode, NN, K_EMB, 1);
  gemm(tnode, HH, in2t, HH, in2b, h, hf16, NN, HH, 0);

  // ---- message passing layers ----
  const size_t edge_smem = (size_t)(128 * K_EDGE + 2 * 128 * HH) * sizeof(_Float16)
                         + (size_t)(128 * 3 + 128) * sizeof(float)
                         + (size_t)128 * sizeof(int);   // ~142 KB < 320 KB/WGP
  for (int l = 0; l < NLAYERS; ++l) {
    egnn_edge_kernel<<<EE / 128, 256, edge_smem, stream>>>(
        eidx, eattr, posc, hf16,
        e1t[l], LP(l, 0), e2t[l], LP(l, 2), x1t[l], LP(l, 10),
        LP(l, 13), LP(l, 12), agg, posacc);
    pack_hcat_kernel<<<(NN * 256 + 255) / 256, 256, 0, stream>>>(hf16, agg, hcat, NN);
    gemm(hcat, K_CAT, h1t[l], K_CAT, LP(l, 4), nullptr, tnode, NN, K_CAT, 1);
    gemm(tnode, HH, h2t[l], HH, LP(l, 6), dh, nullptr, NN, HH, 0);
    layernorm_kernel<<<NN, HH, 0, stream>>>(h, dh, LP(l, 9), LP(l, 8), hf16);
    pos_apply_kernel<<<(NN * 3 + 255) / 256, 256, 0, stream>>>(posc, posacc, NN * 3);
  }

  // ---- readout: out = ro2(silu(ro1(scatter_mean(h)))) ----
  scatter_h_kernel<<<NN, HH, 0, stream>>>(h, batch, hgsum);
  hg_finalize_kernel<<<(GG * HH + 255) / 256, 256, 0, stream>>>(hgsum, cnt, hg16);
  gemm(hg16, HH, ro1t, HH, ro1b, nullptr, tro, GG, HH, 1);
  readout_final_kernel<<<1, 64, 0, stream>>>(tro, ro2w, ro2b, out);
}
